// GalerkinBlock_11003706213292
// MI455X (gfx1250) — compile-verified, matrix operand order checked by reading, not run on hardware
//
#include <hip/hip_runtime.h>
#include <cstdint>
#include <cstddef>

typedef unsigned int u32;
typedef unsigned short u16;

typedef __attribute__((ext_vector_type(16))) __bf16 v16bf;
typedef __attribute__((ext_vector_type(8)))  float  v8f;

#define B_     4
#define N_TOK  8192
#define C_     512
#define H_     4
#define D_     128
#define HID_   2048
#define SCALE_ 0.08838834764831845f   /* D^-0.5 */
#define EPS_   1e-5f

// ---------- scalar helpers ----------
__device__ __forceinline__ u16 f2bf(float f) {
  u32 u = __float_as_uint(f);
  u += 0x7fffu + ((u >> 16) & 1u);            // round-to-nearest-even
  return (u16)(u >> 16);
}
__device__ __forceinline__ float bf2f(u16 h) { return __uint_as_float(((u32)h) << 16); }

__device__ __forceinline__ float wredsum(float v) {
  #pragma unroll
  for (int o = 16; o > 0; o >>= 1) v += __shfl_xor(v, o, 32);
  return v;
}

// ---------- elementwise kernels ----------
__global__ __launch_bounds__(256) void zero_f32_kernel(float* __restrict__ p, long n) {
  long i = (long)blockIdx.x * blockDim.x + threadIdx.x;
  long st = (long)gridDim.x * blockDim.x;
  for (; i < n; i += st) p[i] = 0.0f;
}

// ---------- pack B matrix [K,N] into WMMA-fragment order ----------
// dst layout: [nTile = n/16][kTile = k/32][lane 0..31][dword v 0..7]
//   lane = (n&15) + 16*h ; dword v holds bf16 pair {k, k+1} with
//   k = kTile*32 + 2*(v&3) + 8*h + 16*(v>>2)   (matches 16-bit WMMA B layout)
template <bool FROM_F32>
__global__ __launch_bounds__(256) void pack_b_kernel(const void* __restrict__ src,
                                                     u32* __restrict__ dst,
                                                     int K, int N, int ldb,
                                                     long bsIn, long bsOut) {
  const int z = blockIdx.z;
  const float* sf = (const float*)src + (size_t)z * bsIn;
  const u16*   sh = (const u16*)src + (size_t)z * bsIn;
  u32* d = dst + (size_t)z * bsOut;
  const long kTiles = K >> 5;
  const long total = (long)(K >> 1) * N;            // dwords per batch
  long i = (long)blockIdx.x * blockDim.x + threadIdx.x;
  long st = (long)gridDim.x * blockDim.x;
  for (; i < total; i += st) {
    long nTile = i / (kTiles * 256);
    long rem   = i - nTile * (kTiles * 256);
    int  kTile = (int)(rem >> 8);
    int  w     = (int)(rem & 255);
    int  lane = w >> 3, v = w & 7;
    int  h = lane >> 4, nn = lane & 15;
    int  n = (int)nTile * 16 + nn;
    int  k = kTile * 32 + 2 * (v & 3) + 8 * h + 16 * (v >> 2);
    u32 lo, hi;
    if constexpr (FROM_F32) {
      lo = f2bf(sf[(size_t)k * ldb + n]);
      hi = f2bf(sf[(size_t)(k + 1) * ldb + n]);
    } else {
      lo = sh[(size_t)k * ldb + n];
      hi = sh[(size_t)(k + 1) * ldb + n];
    }
    d[i] = lo | (hi << 16);
  }
}

// ---------- LayerNorm over rows of length 512 (wave per row) ----------
__global__ __launch_bounds__(256) void ln512_kernel(const float* __restrict__ x,
                                                    const float* __restrict__ g,
                                                    const float* __restrict__ b,
                                                    u16* __restrict__ out, int nrows) {
  int wave = threadIdx.x >> 5, lane = threadIdx.x & 31;
  int row = blockIdx.x * 8 + wave;
  if (row >= nrows) return;
  const float* xr = x + (size_t)row * C_;
  float xv[16]; float s = 0.f, sq = 0.f;
  #pragma unroll
  for (int i = 0; i < 16; ++i) { float v = xr[lane + 32 * i]; xv[i] = v; s += v; sq += v * v; }
  s = wredsum(s); sq = wredsum(sq);
  float mu = s * (1.0f / C_);
  float var = sq * (1.0f / C_) - mu * mu;
  float rs = rsqrtf(var + EPS_);
  u16* orow = out + (size_t)row * C_;
  #pragma unroll
  for (int i = 0; i < 16; ++i) { int c = lane + 32 * i; orow[c] = f2bf((xv[i] - mu) * rs * g[c] + b[c]); }
}

// ---------- per-head LN of K (write transposed [bh,D,N]) and V (in place) ----------
__global__ __launch_bounds__(256) void lnkv_kernel(const u16* __restrict__ kin,
                                                   u16* __restrict__ ktout,
                                                   u16* __restrict__ vio,
                                                   const float* __restrict__ gk, const float* __restrict__ bk,
                                                   const float* __restrict__ gv, const float* __restrict__ bv) {
  int wave = threadIdx.x >> 5, lane = threadIdx.x & 31;
  long row = (long)blockIdx.x * 8 + wave;        // [0, B*H*N)
  int bh = (int)(row >> 13);
  int n  = (int)(row & (N_TOK - 1));
  size_t base = ((size_t)bh * N_TOK + n) * D_;
  { // K row -> LN -> transposed scatter (A-operand of the context GEMM)
    float xv[4]; float s = 0.f, sq = 0.f;
    #pragma unroll
    for (int i = 0; i < 4; ++i) { float v = bf2f(kin[base + lane + 32 * i]); xv[i] = v; s += v; sq += v * v; }
    s = wredsum(s); sq = wredsum(sq);
    float mu = s * (1.0f / D_), var = sq * (1.0f / D_) - mu * mu, rs = rsqrtf(var + EPS_);
    #pragma unroll
    for (int i = 0; i < 4; ++i) {
      int d = lane + 32 * i;
      ktout[((size_t)bh * D_ + d) * N_TOK + n] = f2bf((xv[i] - mu) * rs * gk[d] + bk[d]);
    }
  }
  { // V row -> LN in place
    float xv[4]; float s = 0.f, sq = 0.f;
    #pragma unroll
    for (int i = 0; i < 4; ++i) { float v = bf2f(vio[base + lane + 32 * i]); xv[i] = v; s += v; sq += v * v; }
    s = wredsum(s); sq = wredsum(sq);
    float mu = s * (1.0f / D_), var = sq * (1.0f / D_) - mu * mu, rs = rsqrtf(var + EPS_);
    #pragma unroll
    for (int i = 0; i < 4; ++i) { int d = lane + 32 * i; vio[base + d] = f2bf((xv[i] - mu) * rs * gv[d] + bv[d]); }
  }
}

// ---------- tiled bf16 WMMA GEMM ----------
// Block tile BM=128 x BN=64, BK=32; 256 threads = 8 waves, wave w owns rows
// [16w,16w+16) x 64 cols = 4 v8f accumulators.
// A: staged through a PER-WAVE private LDS region (no cross-wave sharing =>
//    no barriers; LDS ops are in-order within a wave). Fragment = 2x ds_load_b128.
// B: pre-packed in WMMA-fragment order => fragment = 2x global_load_b128,
//    pointer-increment addressing only.
enum { EPI_F32R = 0, EPI_QKV = 1, EPI_GELU = 2, EPI_CTX = 3, EPI_ATTN = 4 };

union FragBF { v16bf v; uint4 q[2]; u32 u[8]; };

template <int EPI>
__global__ __launch_bounds__(256) void gemm_bf16_kernel(
    const u16* __restrict__ Ag, const u32* __restrict__ Bp,
    const float* __restrict__ bias, const float* __restrict__ resid,
    float* __restrict__ outf, u16* __restrict__ outb,
    u16* __restrict__ qo, u16* __restrict__ ko, u16* __restrict__ vo,
    int K, int lda, int ldc, int kTilesTotal,
    long bsA, long bsB, long bsO, int splitk, float scale) {
  __shared__ u16 As[8 * 512];                  // 8 waves x (16 rows x 32 cols)

  const int z = blockIdx.z;
  int kTileStart = 0;
  if constexpr (EPI == EPI_CTX) {
    int bh = z / splitk, ks = z % splitk;      // split-K over the N=8192 reduction
    Ag += (long)bh * bsA + (long)ks * K;       // K = per-chunk extent
    Bp += (long)bh * bsB;
    outf += (long)bh * bsO;
    kTileStart = ks * (K >> 5);
  } else {
    Ag += (long)z * bsA;
    Bp += (long)z * bsB;
    if (outf) outf += (long)z * bsO;
  }

  const int m0 = blockIdx.y * 128;
  const int wave = threadIdx.x >> 5;
  const int lane = threadIdx.x & 31;
  const int half = lane >> 4;
  const int l16  = lane & 15;

  u16* Asw = As + wave * 512;

  // strength-reduced source pointers
  const u16* aSrc0 = Ag + (size_t)(m0 + 16 * wave + (lane >> 2)) * lda + (lane & 3) * 8;
  const u16* aSrc1 = aSrc0 + (size_t)8 * lda;
  const u32* bptr0 = Bp + ((size_t)(blockIdx.x * 4 + 0) * kTilesTotal + kTileStart) * 256 + lane * 8;
  const u32* bptr1 = Bp + ((size_t)(blockIdx.x * 4 + 1) * kTilesTotal + kTileStart) * 256 + lane * 8;
  const u32* bptr2 = Bp + ((size_t)(blockIdx.x * 4 + 2) * kTilesTotal + kTileStart) * 256 + lane * 8;
  const u32* bptr3 = Bp + ((size_t)(blockIdx.x * 4 + 3) * kTilesTotal + kTileStart) * 256 + lane * 8;

  const int stIdx0 = lane * 8;                 // LDS store slots for this lane
  const int stIdx1 = lane * 8 + 256;
  const int ldOff  = l16 * 32 + 8 * half;      // A fragment base (u16 units)

  v8f acc[4];
  #pragma unroll
  for (int t = 0; t < 4; ++t)
    #pragma unroll
    for (int r = 0; r < 8; ++r) acc[t][r] = 0.0f;

  const int ktiles = K >> 5;
  for (int kt = 0; kt < ktiles; ++kt) {
    // stage this wave's 16x32 A sub-tile (1KB) into its private LDS region
    *(uint4*)&Asw[stIdx0] = *(const uint4*)aSrc0;
    *(uint4*)&Asw[stIdx1] = *(const uint4*)aSrc1;
    if (kt + 1 < ktiles) {                      // gfx1250 global_prefetch_b8
      __builtin_prefetch(aSrc0 + 32, 0, 1);
      __builtin_prefetch(bptr0 + 256, 0, 1);
    }
    aSrc0 += 32; aSrc1 += 32;

    FragBF a;
    a.q[0] = *(const uint4*)&Asw[ldOff];
    a.q[1] = *(const uint4*)&Asw[ldOff + 16];

    FragBF b0, b1, b2, b3;
    b0.q[0] = *(const uint4*)(bptr0);  b0.q[1] = *(const uint4*)(bptr0 + 4);
    b1.q[0] = *(const uint4*)(bptr1);  b1.q[1] = *(const uint4*)(bptr1 + 4);
    b2.q[0] = *(const uint4*)(bptr2);  b2.q[1] = *(const uint4*)(bptr2 + 4);
    b3.q[0] = *(const uint4*)(bptr3);  b3.q[1] = *(const uint4*)(bptr3 + 4);
    bptr0 += 256; bptr1 += 256; bptr2 += 256; bptr3 += 256;

    acc[0] = __builtin_amdgcn_wmma_f32_16x16x32_bf16(false, a.v, false, b0.v, (short)0, acc[0], false, false);
    acc[1] = __builtin_amdgcn_wmma_f32_16x16x32_bf16(false, a.v, false, b1.v, (short)0, acc[1], false, false);
    acc[2] = __builtin_amdgcn_wmma_f32_16x16x32_bf16(false, a.v, false, b2.v, (short)0, acc[2], false, false);
    acc[3] = __builtin_amdgcn_wmma_f32_16x16x32_bf16(false, a.v, false, b3.v, (short)0, acc[3], false, false);
  }

  // ---- epilogue: C/D layout => col = lane&15, row = vgpr + 8*(lane>>4) ----
  const int n0 = blockIdx.x * 64;
  #pragma unroll
  for (int t = 0; t < 4; ++t) {
    const int j = n0 + t * 16 + l16;
    #pragma unroll
    for (int r = 0; r < 8; ++r) {
      const int m = m0 + wave * 16 + r + 8 * half;
      float val = acc[t][r];
      if constexpr (EPI == EPI_F32R) {
        val += bias[j] + resid[(size_t)m * ldc + j];
        outf[(size_t)m * ldc + j] = val;
      } else if constexpr (EPI == EPI_QKV) {
        val += bias[j];
        int i3 = j >> 9, rem = j & 511, h = rem >> 7, d = rem & 127;
        int bb = m >> 13, n = m & (N_TOK - 1);
        size_t o = (((size_t)(bb * H_ + h)) * N_TOK + n) * D_ + d;
        u16 hv = f2bf(val);
        if (i3 == 0) qo[o] = hv; else if (i3 == 1) ko[o] = hv; else vo[o] = hv;
      } else if constexpr (EPI == EPI_GELU) {
        val += bias[j];
        val = 0.5f * val * (1.0f + erff(val * 0.7071067811865476f));
        outb[(size_t)m * ldc + j] = f2bf(val);
      } else if constexpr (EPI == EPI_CTX) {
        atomicAdd(&outf[(size_t)m * ldc + j], val);   // split-K reduce
      } else { // EPI_ATTN: attn[(b*N+m), h*D + j] bf16, scaled
        int bb = z / H_, h = z % H_;
        size_t o = ((size_t)(bb * N_TOK + m)) * C_ + h * D_ + j;
        outb[o] = f2bf(val * scale);
      }
    }
  }
}

// ---------- host launch ----------
extern "C" void kernel_launch(void* const* d_in, const int* in_sizes, int n_in,
                              void* d_out, int out_size, void* d_ws, size_t ws_size,
                              hipStream_t stream) {
  (void)in_sizes; (void)n_in; (void)out_size; (void)ws_size;
  const float* x      = (const float*)d_in[0];
  const float* n1w    = (const float*)d_in[1];
  const float* n1b    = (const float*)d_in[2];
  const float* qkv_w  = (const float*)d_in[3];
  const float* qkv_b  = (const float*)d_in[4];
  const float* lnk_w  = (const float*)d_in[5];
  const float* lnk_b  = (const float*)d_in[6];
  const float* lnv_w  = (const float*)d_in[7];
  const float* lnv_b  = (const float*)d_in[8];
  const float* proj_w = (const float*)d_in[9];
  const float* proj_b = (const float*)d_in[10];
  const float* n2w    = (const float*)d_in[11];
  const float* n2b    = (const float*)d_in[12];
  const float* mlp_w1 = (const float*)d_in[13];
  const float* mlp_b1 = (const float*)d_in[14];
  const float* mlp_w2 = (const float*)d_in[15];
  const float* mlp_b2 = (const float*)d_in[16];

  char* ws = (char*)d_ws;
  size_t off = 0;
  auto carve = [&](size_t bytes) { void* p = ws + off; off += (bytes + 255) & ~(size_t)255; return p; };
  const long TOK = (long)B_ * N_TOK;                      // 32768
  u32*   WqkvP = (u32*)carve((size_t)C_ * 3 * C_ * 2);    // packed bf16 pairs
  u32*   WprojP= (u32*)carve((size_t)C_ * C_ * 2);
  u32*   Wm1P  = (u32*)carve((size_t)C_ * HID_ * 2);
  u32*   Wm2P  = (u32*)carve((size_t)HID_ * C_ * 2);
  u16*   XN    = (u16*)carve((size_t)TOK * C_ * 2);
  u16*   Qb    = (u16*)carve((size_t)TOK * C_ * 2);       // [bh, n, d]
  u16*   Kb    = (u16*)carve((size_t)TOK * C_ * 2);       // [bh, n, d]
  u16*   Vb    = (u16*)carve((size_t)TOK * C_ * 2);       // [bh, n, d], LN'd in place
  u16*   KT    = (u16*)carve((size_t)TOK * C_ * 2);       // [bh, d, n]
  u32*   Vp    = (u32*)carve((size_t)TOK * C_ * 2);       // V packed per bh
  float* CTX   = (float*)carve((size_t)B_ * H_ * D_ * D_ * 4);
  u32*   CTXBp = (u32*)carve((size_t)B_ * H_ * D_ * D_ * 2);
  u16*   ATTN  = (u16*)carve((size_t)TOK * C_ * 2);       // [token, C]
  float* X1    = (float*)carve((size_t)TOK * C_ * 4);
  u16*   H2    = (u16*)carve((size_t)TOK * C_ * 2);
  u16*   HIDb  = (u16*)carve((size_t)TOK * HID_ * 2);

  // 0) pack weights (f32 -> bf16 fragment order); reused by every M-tile
  pack_b_kernel<true><<<512, 256, 0, stream>>>(qkv_w,  WqkvP, C_, 3 * C_, 3 * C_, 0, 0);
  pack_b_kernel<true><<<256, 256, 0, stream>>>(proj_w, WprojP, C_, C_, C_, 0, 0);
  pack_b_kernel<true><<<512, 256, 0, stream>>>(mlp_w1, Wm1P, C_, HID_, HID_, 0, 0);
  pack_b_kernel<true><<<512, 256, 0, stream>>>(mlp_w2, Wm2P, HID_, C_, C_, 0, 0);

  // 1) LN1 -> bf16
  ln512_kernel<<<TOK / 8, 256, 0, stream>>>(x, n1w, n1b, XN, (int)TOK);

  // 2) QKV GEMM [32768,512]x[512,1536], per-head bf16 scatter epilogue
  gemm_bf16_kernel<EPI_QKV><<<dim3(3 * C_ / 64, TOK / 128, 1), 256, 0, stream>>>(
      XN, WqkvP, qkv_b, nullptr, nullptr, nullptr, Qb, Kb, Vb,
      C_, C_, 3 * C_, C_ / 32, 0, 0, 0, 1, 1.0f);

  // 3) per-head LN of K (transposed out) and V (in place)
  lnkv_kernel<<<(B_ * H_ * N_TOK) / 8, 256, 0, stream>>>(Kb, KT, Vb, lnk_w, lnk_b, lnv_w, lnv_b);

  // 4) pack V per (b,h) as context-GEMM B operand
  pack_b_kernel<false><<<dim3(1024, 1, B_ * H_), 256, 0, stream>>>(
      Vb, Vp, N_TOK, D_, D_, (long)N_TOK * D_, (long)N_TOK * D_ / 2);

  // 5) context = K^T V per (b,h): [128,8192]x[8192,128], split-K=16, atomic f32
  zero_f32_kernel<<<256, 256, 0, stream>>>(CTX, (long)B_ * H_ * D_ * D_);
  const int SPLITK = 16, KCHUNK = N_TOK / SPLITK;          // 512
  gemm_bf16_kernel<EPI_CTX><<<dim3(D_ / 64, 1, B_ * H_ * SPLITK), 256, 0, stream>>>(
      KT, Vp, nullptr, nullptr, CTX, nullptr, nullptr, nullptr, nullptr,
      KCHUNK, N_TOK, D_, N_TOK / 32,
      (long)D_ * N_TOK, (long)N_TOK * D_ / 2, (long)D_ * D_, SPLITK, 1.0f);

  // 6) pack context f32 -> bf16 fragment order per (b,h)
  pack_b_kernel<true><<<dim3(32, 1, B_ * H_), 256, 0, stream>>>(
      CTX, CTXBp, D_, D_, D_, (long)D_ * D_, (long)D_ * D_ / 2);

  // 7) attn = Q @ ctx * SCALE per (b,h): [8192,128]x[128,128] -> bf16 [token,C]
  gemm_bf16_kernel<EPI_ATTN><<<dim3(D_ / 64, N_TOK / 128, B_ * H_), 256, 0, stream>>>(
      Qb, CTXBp, nullptr, nullptr, nullptr, ATTN, nullptr, nullptr, nullptr,
      D_, D_, D_, D_ / 32,
      (long)N_TOK * D_, (long)D_ * D_ / 2, 0, 1, SCALE_);

  // 8) x1 = shortcut + attn @ proj_w + proj_b  (f32)
  gemm_bf16_kernel<EPI_F32R><<<dim3(C_ / 64, TOK / 128, 1), 256, 0, stream>>>(
      ATTN, WprojP, proj_b, x, X1, nullptr, nullptr, nullptr, nullptr,
      C_, C_, C_, C_ / 32, 0, 0, 0, 1, 1.0f);

  // 9) LN2 -> bf16
  ln512_kernel<<<TOK / 8, 256, 0, stream>>>(X1, n2w, n2b, H2, (int)TOK);

  // 10) hidden = gelu(h @ mlp_w1 + b1) -> bf16 [32768,2048]
  gemm_bf16_kernel<EPI_GELU><<<dim3(HID_ / 64, TOK / 128, 1), 256, 0, stream>>>(
      H2, Wm1P, mlp_b1, nullptr, nullptr, HIDb, nullptr, nullptr, nullptr,
      C_, C_, HID_, C_ / 32, 0, 0, 0, 1, 1.0f);

  // 11) out = x1 + hidden @ mlp_w2 + b2  (f32, final output)
  gemm_bf16_kernel<EPI_F32R><<<dim3(C_ / 64, TOK / 128, 1), 256, 0, stream>>>(
      HIDb, Wm2P, mlp_b2, X1, (float*)d_out, nullptr, nullptr, nullptr, nullptr,
      HID_, HID_, C_, HID_ / 32, 0, 0, 0, 1, 1.0f);
}
